// DistributedAttentionS2_65592740545205
// MI455X (gfx1250) — matrix operand order repaired and checked by reading, not compiled, not run on hardware
//
#include <hip/hip_runtime.h>
#include <hip/hip_bf16.h>
#include <stdint.h>

// ---------------------------------------------------------------------------
// Spherical (S2) multi-head self-attention for MI455X / gfx1250, wave32 WMMA.
// B=1, C=256, H=46, W=90 -> N=4140 (padded to 4160), HEADS=8, d=32.
// All GEMMs use v_wmma_f32_16x16x32_f16 (f32 accumulate, f16 operands).
// K/V tiles in the attention loop are staged LDS-side with
// GLOBAL_LOAD_ASYNC_TO_LDS_B128 + s_wait_asynccnt double buffering when the
// toolchain exposes the builtin (fallback: direct L2-resident fragment loads).
// ---------------------------------------------------------------------------

typedef __attribute__((ext_vector_type(16))) _Float16 v16h;
typedef __attribute__((ext_vector_type(8)))  float    v8f;
typedef int v4i __attribute__((vector_size(16)));   // builtin's pointee type

#define NSEQ   4140
#define NPAD   4160          // multiple of 32
#define CIN    256
#define NHEAD  8
#define DHEAD  32
#define NKT    (NPAD / 32)   // 130 key tiles

#if defined(__has_builtin)
#if __has_builtin(__builtin_amdgcn_global_load_async_to_lds_b128)
#define USE_ASYNC_LDS 1
#endif
#endif
#ifndef USE_ASYNC_LDS
#define USE_ASYNC_LDS 0
#endif

union HFrag  { v16h v; uint4 q[2]; _Float16 h[16]; };
union HPack8 { uint4 q;            _Float16 h[8];  };

#if USE_ASYNC_LDS
template <int N>
__device__ __forceinline__ void wait_async()
{
#if __has_builtin(__builtin_amdgcn_s_wait_asynccnt)
    __builtin_amdgcn_s_wait_asynccnt(N);
#else
    asm volatile("s_wait_asynccnt %0" :: "i"(N) : "memory");
#endif
}

__device__ __forceinline__ void async_cp16(const _Float16* g, _Float16* l)
{
    __builtin_amdgcn_global_load_async_to_lds_b128(
        (__attribute__((address_space(1))) v4i*)(const void*)g,
        (__attribute__((address_space(3))) v4i*)(void*)l,
        0, 0);
}
#endif

// ============================================================================
// Kernel 1: fused Q/K/V projection (1x1 conv == GEMM), f32 -> f16 head-packed.
//   Out[k][n] = sum_c W[k][c] * X[c][n] + b[k]
//   A = W tile (16k x 32c), B = X tile (32c x 16n) via LDS transpose.
//   qh, kh : [head][n][32]  (d-contiguous)   vt : [head][32][n] (n-contiguous)
// ============================================================================
__global__ __launch_bounds__(128)
void qkv_proj_kernel(const float* __restrict__ X,
                     const float* __restrict__ qw, const float* __restrict__ qb,
                     const float* __restrict__ kw, const float* __restrict__ kb,
                     const float* __restrict__ vw, const float* __restrict__ vb,
                     _Float16* __restrict__ qh, _Float16* __restrict__ kh,
                     _Float16* __restrict__ vt)
{
    __shared__ _Float16 Xs[64][40];        // [n][c] tile, stride 40 -> 16B align
    __shared__ _Float16 Ws[3][32][40];     // [mat][k][c] tiles

    const int tid  = threadIdx.x;
    const int lane = tid & 31;
    const int wave = tid >> 5;
    const int lo   = lane & 15;
    const int hi   = lane >> 4;

    const int n0   = blockIdx.x * 64;      // 65 blocks cover NPAD
    const int head = blockIdx.y;

    const float* Wm[3] = { qw, kw, vw };

    // X staging plan: thread owns one n (predicate hoisted, mask-multiply,
    // clamped address -> branch-free) and 16 consecutive c values.
    const int nn    = tid & 63;
    const int cseg  = (tid >> 6) * 16;     // 0 or 16
    const int nglb  = n0 + nn;
    const float xm  = (nglb < NSEQ) ? 1.0f : 0.0f;
    const int nsafe = (nglb < NSEQ) ? nglb : (NSEQ - 1);

    // W staging plan: thread owns one k row, 8 consecutive c values.
    const int wkk = tid >> 2;              // 0..31
    const int wcb = (tid & 3) * 8;         // 0,8,16,24

    v8f zero = {};
    v8f acc[3][2];
#pragma unroll
    for (int m = 0; m < 3; ++m)
#pragma unroll
        for (int s = 0; s < 2; ++s) acc[m][s] = zero;

    for (int cs = 0; cs < CIN / 32; ++cs) {
        const int c0 = cs * 32;

        // ---- stage X tile (transpose [c][n] -> [n][c], f32 -> f16) ----
        {
            HFrag xp;
#pragma unroll
            for (int i = 0; i < 16; ++i)
                xp.h[i] = (_Float16)(X[(size_t)(c0 + cseg + i) * NSEQ + nsafe] * xm);
            *reinterpret_cast<uint4*>(&Xs[nn][cseg])     = xp.q[0];
            *reinterpret_cast<uint4*>(&Xs[nn][cseg + 8]) = xp.q[1];
        }
        // ---- stage the three weight tiles (one b128 store per thread) ----
#pragma unroll
        for (int m = 0; m < 3; ++m) {
            const float* wr = Wm[m] + (size_t)(head * 32 + wkk) * CIN + c0 + wcb;
            HPack8 wp;
#pragma unroll
            for (int j = 0; j < 8; ++j) wp.h[j] = (_Float16)wr[j];
            *reinterpret_cast<uint4*>(&Ws[m][wkk][wcb]) = wp.q;
        }
        __syncthreads();

        // B fragment: col n = wave*16 + lo, K(=c) = g*16 + hi*8 + j (table)
        HFrag bf;
        {
            const _Float16* xr = &Xs[wave * 16 + lo][hi * 8];
            bf.q[0] = *reinterpret_cast<const uint4*>(xr);
            bf.q[1] = *reinterpret_cast<const uint4*>(xr + 16);
        }
#pragma unroll
        for (int m = 0; m < 3; ++m) {
#pragma unroll
            for (int s = 0; s < 2; ++s) {
                // A fragment: row k = s*16+lo, K(=c) = g*16 + hi*8 + w
                HFrag af;
                const _Float16* wr = &Ws[m][s * 16 + lo][0];
                af.q[0] = *reinterpret_cast<const uint4*>(wr + hi * 8);
                af.q[1] = *reinterpret_cast<const uint4*>(wr + 16 + hi * 8);
                acc[m][s] = __builtin_amdgcn_wmma_f32_16x16x32_f16(
                    false, af.v, false, bf.v, (short)0, acc[m][s], false, false);
            }
        }
        __syncthreads();
    }

    // ---- epilogue: D layout col n = lo, rows k = r + 8*hi within sub-tile ----
    const int ncol = n0 + wave * 16 + lo;
#pragma unroll
    for (int s = 0; s < 2; ++s) {
        {   // Q: [head][n][d], packed 8 contiguous halves per lane
            HPack8 pk;
#pragma unroll
            for (int r = 0; r < 8; ++r)
                pk.h[r] = (_Float16)(acc[0][s][r] +
                                     qb[head * 32 + s * 16 + hi * 8 + r]);
            *reinterpret_cast<uint4*>(
                &qh[((size_t)(head * NPAD + ncol)) * 32 + s * 16 + hi * 8]) = pk.q;
        }
        {   // K: same layout
            HPack8 pk;
#pragma unroll
            for (int r = 0; r < 8; ++r)
                pk.h[r] = (_Float16)(acc[1][s][r] +
                                     kb[head * 32 + s * 16 + hi * 8 + r]);
            *reinterpret_cast<uint4*>(
                &kh[((size_t)(head * NPAD + ncol)) * 32 + s * 16 + hi * 8]) = pk.q;
        }
        // V transposed: vt[head][d][n]
#pragma unroll
        for (int r = 0; r < 8; ++r) {
            int d = s * 16 + hi * 8 + r;
            vt[((size_t)(head * 32 + d)) * NPAD + ncol] =
                (_Float16)(acc[2][s][r] + vb[head * 32 + d]);
        }
    }
}

// ============================================================================
// Kernel 2: flash attention per head. Block = 1 head x 32 query rows (2 waves).
// scores = (Q K^T) * scale + log_qw[key]; online softmax; out = P V.
// ============================================================================
__global__ __launch_bounds__(64)
void attn_kernel(const _Float16* __restrict__ qh, const _Float16* __restrict__ kh,
                 const _Float16* __restrict__ vt, const float* __restrict__ lqw,
                 _Float16* __restrict__ attn)
{
    __shared__ _Float16 Ps[2][16][40];     // per-wave P transpose staging
#if USE_ASYNC_LDS
    __shared__ _Float16 Ks[2][32 * 32];    // double-buffered K tile [key][d]
    __shared__ _Float16 Vs[2][32 * 32];    // double-buffered V tile [d][key]
#endif

    const int tid  = threadIdx.x;
    const int lane = tid & 31;
    const int wave = tid >> 5;
    const int lo   = lane & 15;
    const int hi   = lane >> 4;
    const int head = blockIdx.y;
    const int q0   = blockIdx.x * 32 + wave * 16;

    const float scale = 0.17677669529663687f;   // 1/sqrt(32)

    const _Float16* kh_head = kh + (size_t)head * NPAD * 32;
    const _Float16* vt_head = vt + (size_t)head * DHEAD * NPAD;

    // Q A-fragment: row q = q0+lo, K(=d) contiguous in qh
    HFrag qf;
    {
        const _Float16* qr = qh + ((size_t)(head * NPAD + q0 + lo)) * 32;
        qf.q[0] = *reinterpret_cast<const uint4*>(qr + hi * 16);
        qf.q[1] = *reinterpret_cast<const uint4*>(qr + hi * 16 + 8);
    }

    float mrow[8], lrow[8];
#pragma unroll
    for (int r = 0; r < 8; ++r) { mrow[r] = -3.0e38f; lrow[r] = 0.0f; }
    v8f acc0 = {}, acc1 = {};

#if USE_ASYNC_LDS
    // async staging plan: thread copies 2x16B of K tile and 2x16B of V tile.
    const int ch0 = tid * 2;
    auto stage = [&](int k0, int buf) {
#pragma unroll
        for (int j = 0; j < 2; ++j) {
            int ch  = ch0 + j;
            int row = ch >> 2;              // 0..31
            int off = (ch & 3) * 8;         // halves
            async_cp16(kh_head + (size_t)(k0 + row) * 32 + off,
                       &Ks[buf][row * 32 + off]);
            async_cp16(vt_head + (size_t)row * NPAD + k0 + off,
                       &Vs[buf][row * 32 + off]);
        }
    };
    stage(0, 0);                            // preload tile 0 (4 asyncs/wave)
#endif

    for (int kt = 0; kt < NKT; ++kt) {
        const int k0  = kt * 32;
#if USE_ASYNC_LDS
        const int buf = kt & 1;
        if (kt + 1 < NKT) {                 // keep next tile in flight
            stage((kt + 1) * 32, buf ^ 1);
            wait_async<4>();                // current tile's 4 asyncs landed
        } else {
            wait_async<0>();
        }
        __syncthreads();                    // both waves' copies visible

        HFrag kf0, kf1;
        {
            const _Float16* k0p = &Ks[buf][(lo) * 32 + hi * 16];
            kf0.q[0] = *reinterpret_cast<const uint4*>(k0p);
            kf0.q[1] = *reinterpret_cast<const uint4*>(k0p + 8);
            const _Float16* k1p = &Ks[buf][(16 + lo) * 32 + hi * 16];
            kf1.q[0] = *reinterpret_cast<const uint4*>(k1p);
            kf1.q[1] = *reinterpret_cast<const uint4*>(k1p + 8);
        }
#else
        if (kt + 1 < NKT)
            __builtin_prefetch(kh_head + ((size_t)(k0 + 32 + lo)) * 32, 0, 1);
        HFrag kf0, kf1;
        {
            const _Float16* k0p = kh_head + ((size_t)(k0 + lo)) * 32 + hi * 16;
            kf0.q[0] = *reinterpret_cast<const uint4*>(k0p);
            kf0.q[1] = *reinterpret_cast<const uint4*>(k0p + 8);
            const _Float16* k1p = kh_head + ((size_t)(k0 + 16 + lo)) * 32 + hi * 16;
            kf1.q[0] = *reinterpret_cast<const uint4*>(k1p);
            kf1.q[1] = *reinterpret_cast<const uint4*>(k1p + 8);
        }
#endif
        v8f zc = {};
        v8f s0 = __builtin_amdgcn_wmma_f32_16x16x32_f16(
            false, qf.v, false, kf0.v, (short)0, zc, false, false);
        v8f s1 = __builtin_amdgcn_wmma_f32_16x16x32_f16(
            false, qf.v, false, kf1.v, (short)0, zc, false, false);

        const int key0 = k0 + lo;
        const int key1 = k0 + 16 + lo;
        const float b0 = (key0 < NSEQ) ? lqw[key0] : -1.0e30f;
        const float b1 = (key1 < NSEQ) ? lqw[key1] : -1.0e30f;

        float p0[8], p1[8], sc[8];
#pragma unroll
        for (int r = 0; r < 8; ++r) {
            float v0 = s0[r] * scale + b0;
            float v1 = s1[r] * scale + b1;
            float cur = fmaxf(v0, v1);
#pragma unroll
            for (int m = 1; m < 16; m <<= 1)
                cur = fmaxf(cur, __shfl_xor(cur, m, 32));
            float nm = fmaxf(mrow[r], cur);
            float so = __expf(mrow[r] - nm);
            float e0 = __expf(v0 - nm);
            float e1 = __expf(v1 - nm);
            float rs = e0 + e1;
#pragma unroll
            for (int m = 1; m < 16; m <<= 1)
                rs += __shfl_xor(rs, m, 32);
            lrow[r] = lrow[r] * so + rs;
            mrow[r] = nm;
            sc[r] = so;  p0[r] = e0;  p1[r] = e1;
        }
#pragma unroll
        for (int r = 0; r < 8; ++r) { acc0[r] *= sc[r]; acc1[r] *= sc[r]; }

        // P: D layout -> A layout through LDS
#pragma unroll
        for (int r = 0; r < 8; ++r) {
            Ps[wave][r + 8 * hi][lo]      = (_Float16)p0[r];
            Ps[wave][r + 8 * hi][16 + lo] = (_Float16)p1[r];
        }
        __syncthreads();
        HFrag pf;
        {
            const _Float16* pr = &Ps[wave][lo][0];
            pf.q[0] = *reinterpret_cast<const uint4*>(pr + hi * 16);
            pf.q[1] = *reinterpret_cast<const uint4*>(pr + hi * 16 + 8);
        }
        HFrag vf0, vf1;
#if USE_ASYNC_LDS
        {
            const _Float16* v0p = &Vs[buf][(lo) * 32 + hi * 16];
            vf0.q[0] = *reinterpret_cast<const uint4*>(v0p);
            vf0.q[1] = *reinterpret_cast<const uint4*>(v0p + 8);
            const _Float16* v1p = &Vs[buf][(16 + lo) * 32 + hi * 16];
            vf1.q[0] = *reinterpret_cast<const uint4*>(v1p);
            vf1.q[1] = *reinterpret_cast<const uint4*>(v1p + 8);
        }
#else
        {
            const _Float16* v0p = vt_head + ((size_t)lo) * NPAD + k0 + hi * 16;
            vf0.q[0] = *reinterpret_cast<const uint4*>(v0p);
            vf0.q[1] = *reinterpret_cast<const uint4*>(v0p + 8);
            const _Float16* v1p = vt_head + ((size_t)(16 + lo)) * NPAD + k0 + hi * 16;
            vf1.q[0] = *reinterpret_cast<const uint4*>(v1p);
            vf1.q[1] = *reinterpret_cast<const uint4*>(v1p + 8);
        }
#endif
        acc0 = __builtin_amdgcn_wmma_f32_16x16x32_f16(
            false, pf.v, false, vf0.v, (short)0, acc0, false, false);
        acc1 = __builtin_amdgcn_wmma_f32_16x16x32_f16(
            false, pf.v, false, vf1.v, (short)0, acc1, false, false);
        __syncthreads();
    }

    // epilogue: normalize rows, store attn[n][head*32 + d] as f16
#pragma unroll
    for (int r = 0; r < 8; ++r) {
        float inv = 1.0f / lrow[r];
        int q = q0 + r + 8 * hi;
        _Float16* orow = attn + (size_t)q * 256 + head * 32;
        orow[lo]      = (_Float16)(acc0[r] * inv);
        orow[16 + lo] = (_Float16)(acc1[r] * inv);
    }
}

// ============================================================================
// Kernel 3: output projection. out[k][n] = sum_c attn[n][c] * pw[k][c] + pb[k]
// Direct global fragment loads (everything is L2-resident).
// ============================================================================
__global__ __launch_bounds__(128)
void proj_out_kernel(const _Float16* __restrict__ attn,
                     const float* __restrict__ pw, const float* __restrict__ pb,
                     float* __restrict__ out)
{
    const int tid  = threadIdx.x;
    const int lane = tid & 31;
    const int wave = tid >> 5;
    const int lo   = lane & 15;
    const int hi   = lane >> 4;
    const int n0   = blockIdx.x * 64 + wave * 16;
    const int k0   = blockIdx.y * 16;

    v8f acc = {};
    for (int cs = 0; cs < CIN / 32; ++cs) {
        const int c0 = cs * 32;
        // A fragment from pw (f32 -> f16 convert on load)
        HFrag af;
        {
            const float* wr = pw + (size_t)(k0 + lo) * CIN + c0;
#pragma unroll
            for (int j = 0; j < 8; ++j) af.h[j]     = (_Float16)wr[hi * 8 + j];
#pragma unroll
            for (int j = 0; j < 8; ++j) af.h[8 + j] = (_Float16)wr[16 + hi * 8 + j];
        }
        // B fragment from attn: col = n, K(=c) = g*16 + hi*8 + j (table)
        HFrag bf;
        {
            const _Float16* ar = attn + (size_t)(n0 + lo) * 256 + c0 + hi * 8;
            bf.q[0] = *reinterpret_cast<const uint4*>(ar);
            bf.q[1] = *reinterpret_cast<const uint4*>(ar + 16);
        }
        acc = __builtin_amdgcn_wmma_f32_16x16x32_f16(
            false, af.v, false, bf.v, (short)0, acc, false, false);
    }

    const int ncol = n0 + lo;
    if (ncol < NSEQ) {
#pragma unroll
        for (int r = 0; r < 8; ++r) {
            int k = k0 + r + 8 * hi;
            out[(size_t)k * NSEQ + ncol] = acc[r] + pb[k];
        }
    }
}

// ============================================================================
extern "C" void kernel_launch(void* const* d_in, const int* in_sizes, int n_in,
                              void* d_out, int out_size, void* d_ws, size_t ws_size,
                              hipStream_t stream)
{
    const float* query = (const float*)d_in[0];
    const float* qw    = (const float*)d_in[1];
    const float* qb    = (const float*)d_in[2];
    const float* kw    = (const float*)d_in[3];
    const float* kb    = (const float*)d_in[4];
    const float* vw    = (const float*)d_in[5];
    const float* vb    = (const float*)d_in[6];
    const float* pw    = (const float*)d_in[7];
    const float* pb    = (const float*)d_in[8];
    const float* lqw   = (const float*)d_in[9];
    float*       out   = (float*)d_out;

    // workspace carve-up (f16 elements)
    _Float16* ws = (_Float16*)d_ws;
    const size_t perM = (size_t)NHEAD * NPAD * DHEAD;   // 1,064,960 halves
    _Float16* qh   = ws;
    _Float16* kh   = ws + perM;
    _Float16* vt   = ws + 2 * perM;
    _Float16* attn = ws + 3 * perM;

    qkv_proj_kernel<<<dim3(NPAD / 64, NHEAD), 128, 0, stream>>>(
        query, qw, qb, kw, kb, vw, vb, qh, kh, vt);
    attn_kernel<<<dim3(NPAD / 32, NHEAD), 64, 0, stream>>>(
        qh, kh, vt, lqw, attn);
    proj_out_kernel<<<dim3(NPAD / 64, CIN / 16), 128, 0, stream>>>(
        attn, pw, pb, out);
    (void)in_sizes; (void)n_in; (void)out_size; (void)ws_size;
}
